// Attention_944892805301
// MI455X (gfx1250) — compile-verified, matrix operand order checked by reading, not run on hardware
//
#include <hip/hip_runtime.h>
#include <hip/hip_bf16.h>
#include <math.h>

typedef __attribute__((ext_vector_type(16))) __bf16          v16bf;
typedef __attribute__((ext_vector_type(16))) unsigned short  v16us;
typedef __attribute__((ext_vector_type(8)))  float           v8f;
typedef int v4i_vs __attribute__((vector_size(16)));   // matches builtin's "V4i"

#define N_HEAD 16
#define DHEAD  64
#define TSEQ   2048
#define EMB    1024
#define BATCH  2

// ---- CDNA5 async global->LDS path (guarded; host pass falls back cleanly) ----
#if defined(__has_builtin)
#if __has_builtin(__builtin_amdgcn_global_load_async_to_lds_b128) && \
    __has_builtin(__builtin_amdgcn_s_wait_asynccnt)
#define HAS_ASYNC_LDS 1
#endif
#endif
#ifndef HAS_ASYNC_LDS
#define HAS_ASYNC_LDS 0
#endif

#if HAS_ASYNC_LDS
#define GASYNC_B128(gptr, lptr)                                         \
  __builtin_amdgcn_global_load_async_to_lds_b128(                       \
      (__attribute__((address_space(1))) v4i_vs*)(gptr),                \
      (__attribute__((address_space(3))) v4i_vs*)(lptr), 0, 0)
#endif

__device__ __forceinline__ unsigned short f2bf(float f) {
  unsigned int u = __builtin_bit_cast(unsigned int, f);
  u = (u + 0x7fffu + ((u >> 16) & 1u)) >> 16;   // round-to-nearest-even
  return (unsigned short)u;
}

__device__ __forceinline__ v8f wmma_bf16(v16bf a, v16bf b, v8f c) {
  // D = A(16x32 bf16) * B(32x16 bf16) + C(16x16 f32)
  return __builtin_amdgcn_wmma_f32_16x16x32_bf16(false, a, false, b, (short)0, c,
                                                 false, false);
}

// 16x32 bf16 tile; this lane's 16 elements come from one contiguous K-row.
// A-operand: lane idx = matrix row. Also valid as B-operand when B's columns
// are stored as contiguous K-rows (memory holds B^T row-major).
// K pattern: lanes 0-15 -> K {0..7, 16..23}; lanes 16-31 -> K {8..15, 24..31}.
__device__ __forceinline__ v16bf load_tile_rowk(const unsigned short* base,
                                                int row_stride) {
  const int lane = threadIdx.x & 31;
  const int half = lane >> 4;
  const int idx  = lane & 15;
  const unsigned short* p = base + (size_t)idx * row_stride + half * 8;
  v16us r;
#pragma unroll
  for (int i = 0; i < 8; ++i) r[i] = p[i];
#pragma unroll
  for (int i = 0; i < 8; ++i) r[8 + i] = p[16 + i];
  return __builtin_bit_cast(v16bf, r);
}

// B-operand from K-major storage: element i = base[k(i)*stride + col]
__device__ __forceinline__ v16bf load_tile_colk(const unsigned short* base,
                                                int row_stride) {
  const int lane = threadIdx.x & 31;
  const int half = lane >> 4;
  const int idx  = lane & 15;
  v16us r;
#pragma unroll
  for (int i = 0; i < 16; ++i) {
    int k = half * 8 + (i < 8 ? i : i + 8);
    r[i] = base[(size_t)k * row_stride + idx];
  }
  return __builtin_bit_cast(v16bf, r);
}

// ---------------- stage kernels ----------------

__global__ void k_f32_to_bf16(const float* __restrict__ in,
                              unsigned short* __restrict__ out, int n) {
  int i = blockIdx.x * blockDim.x + threadIdx.x;
  if (i < n) out[i] = f2bf(in[i]);
}

// out[c][r] = bf16(in[r][c]); in is R x C f32, out is C x R bf16. 32x32 tiles.
__global__ void k_transpose_bf16(const float* __restrict__ in,
                                 unsigned short* __restrict__ out,
                                 int R, int C) {
  __shared__ float tile[32][33];
  const int c0 = blockIdx.x * 32, r0 = blockIdx.y * 32;
  const int tx = threadIdx.x, ty = threadIdx.y;   // 32 x 8
#pragma unroll
  for (int i = 0; i < 4; ++i) {
    int r = ty + i * 8;
    tile[r][tx] = in[(size_t)(r0 + r) * C + (c0 + tx)];
  }
  __syncthreads();
#pragma unroll
  for (int i = 0; i < 4; ++i) {
    int r = ty + i * 8;
    out[(size_t)(c0 + r) * R + (r0 + tx)] = f2bf(tile[tx][r]);
  }
}

// C(MxN f32) = A(MxK bf16 row-major) * Bt(NxK bf16 row-major)^T + bias.
// 256 threads / 8 waves; block tile 128x128; LDS double-buffered A/B tiles
// staged with async global->LDS b128 copies; wave (wr,wc) computes 32x64.
__global__ void __launch_bounds__(256)
k_gemm_bf16(const unsigned short* __restrict__ A,
            const unsigned short* __restrict__ Bt,
            const float* __restrict__ bias,
            float* __restrict__ C, int M, int N, int K) {
  __shared__ unsigned short Abuf[2][128 * 32];
  __shared__ unsigned short Bbuf[2][128 * 32];

  const int tid  = threadIdx.x;
  const int w    = tid >> 5;
  const int lane = tid & 31;
  const int half = lane >> 4, idx = lane & 15;
  const int wr = w & 3;        // 0..3 -> 32-row slice
  const int wc = w >> 2;       // 0..1 -> 64-col slice

  const int blkM = blockIdx.x * 128;
  const int blkN = blockIdx.y * 128;
  const unsigned short* Ag  = A  + (size_t)blkM * K;
  const unsigned short* Btg = Bt + (size_t)blkN * K;

  // tile = 128 rows x 32 bf16 = 512 chunks of 8 ushort (16B); 2 chunks/thread
  const int c0 = tid, c1 = tid + 256;
  const int ar0 = c0 >> 2, ao0 = (c0 & 3) * 8;
  const int ar1 = c1 >> 2, ao1 = (c1 & 3) * 8;

  v8f zero = {};
  v8f acc[2][4];
#pragma unroll
  for (int i = 0; i < 2; ++i)
#pragma unroll
    for (int t = 0; t < 4; ++t) acc[i][t] = zero;

  auto issue = [&](int buf, int k) {
#if HAS_ASYNC_LDS
    GASYNC_B128(Ag  + (size_t)ar0 * K + k + ao0, &Abuf[buf][c0 * 8]);
    GASYNC_B128(Ag  + (size_t)ar1 * K + k + ao1, &Abuf[buf][c1 * 8]);
    GASYNC_B128(Btg + (size_t)ar0 * K + k + ao0, &Bbuf[buf][c0 * 8]);
    GASYNC_B128(Btg + (size_t)ar1 * K + k + ao1, &Bbuf[buf][c1 * 8]);
#else
    uint4 a0 = *(const uint4*)(Ag  + (size_t)ar0 * K + k + ao0);
    uint4 a1 = *(const uint4*)(Ag  + (size_t)ar1 * K + k + ao1);
    uint4 b0 = *(const uint4*)(Btg + (size_t)ar0 * K + k + ao0);
    uint4 b1 = *(const uint4*)(Btg + (size_t)ar1 * K + k + ao1);
    *(uint4*)&Abuf[buf][c0 * 8] = a0;
    *(uint4*)&Abuf[buf][c1 * 8] = a1;
    *(uint4*)&Bbuf[buf][c0 * 8] = b0;
    *(uint4*)&Bbuf[buf][c1 * 8] = b1;
#endif
  };

  const int nk = K / 32;
  issue(0, 0);
  for (int i = 0; i < nk; ++i) {
    if (i + 1 < nk) {
      issue((i + 1) & 1, (i + 1) * 32);   // overlap next tile with this one
#if HAS_ASYNC_LDS
      __builtin_amdgcn_s_wait_asynccnt(4);  // 4 newer in flight -> buf i landed
#endif
    } else {
#if HAS_ASYNC_LDS
      __builtin_amdgcn_s_wait_asynccnt(0);
#endif
    }
    __syncthreads();
    const unsigned short* Ab = &Abuf[i & 1][0];
    const unsigned short* Bb = &Bbuf[i & 1][0];
    v16bf a0 = load_tile_rowk(Ab + (wr * 32 +  0) * 32, 32);
    v16bf a1 = load_tile_rowk(Ab + (wr * 32 + 16) * 32, 32);
#pragma unroll
    for (int t = 0; t < 4; ++t) {
      v16bf b = load_tile_rowk(Bb + (wc * 64 + t * 16) * 32, 32);
      acc[0][t] = wmma_bf16(a0, b, acc[0][t]);
      acc[1][t] = wmma_bf16(a1, b, acc[1][t]);
    }
    __syncthreads();   // all waves done with buf i before it is re-staged
  }

  const int m0 = blkM + wr * 32;
  const int n0 = blkN + wc * 64;
#pragma unroll
  for (int i = 0; i < 2; ++i)
#pragma unroll
    for (int t = 0; t < 4; ++t) {
      int col = n0 + t * 16 + idx;
      float bv = bias ? bias[col] : 0.f;
#pragma unroll
      for (int r = 0; r < 8; ++r) {
        int row = m0 + i * 16 + r + half * 8;   // C layout: vgpr r -> rows r/r+8
        C[(size_t)row * N + col] = acc[i][t][r] + bv;
      }
    }
}

// qkv f32 (B*T x 3E) -> Q,K (RoPE'd, Q pre-scaled by 1/sqrt(D)), V as
// bf16 in (B,H,T,D). One thread per rotation pair j.
__global__ void k_rope_pack(const float* __restrict__ qkv,
                            unsigned short* __restrict__ Q,
                            unsigned short* __restrict__ Ko,
                            unsigned short* __restrict__ V) {
  int flat = blockIdx.x * blockDim.x + threadIdx.x;  // = b|h|t|j bits
  int j = flat & 31;
  int t = (flat >> 5) & (TSEQ - 1);
  int h = (flat >> 16) & (N_HEAD - 1);
  int b = flat >> 20;
  if (b >= BATCH) return;

  const float* row = qkv + (size_t)(b * TSEQ + t) * (3 * EMB);
  float ang = (float)t * powf(10000.0f, -((float)(2 * j) / (float)DHEAD));
  float c = cosf(ang), s = sinf(ang);
  int d0 = 2 * j;
  float q1 = row[h * DHEAD + d0],           q2 = row[h * DHEAD + d0 + 1];
  float k1 = row[EMB + h * DHEAD + d0],     k2 = row[EMB + h * DHEAD + d0 + 1];
  float v1 = row[2 * EMB + h * DHEAD + d0], v2 = row[2 * EMB + h * DHEAD + d0 + 1];
  const float qs = 0.125f;  // 1/sqrt(64), folded into Q
  size_t o = ((size_t)(b * N_HEAD + h) * TSEQ + t) * DHEAD + d0;
  Q[o]      = f2bf((q1 * c - q2 * s) * qs);
  Q[o + 1]  = f2bf((q1 * s + q2 * c) * qs);
  Ko[o]     = f2bf(k1 * c - k2 * s);
  Ko[o + 1] = f2bf(k1 * s + k2 * c);
  V[o]      = f2bf(v1);
  V[o + 1]  = f2bf(v2);
}

// Flash attention: 1 block = 4 waves = 64 query rows of one (b,h);
// KV in 32-wide LDS-staged chunks (async copies); online softmax;
// Y out as bf16 (B,T,E).
__global__ void __launch_bounds__(128)
k_flash_attn(const unsigned short* __restrict__ Q,
             const unsigned short* __restrict__ Kg,
             const unsigned short* __restrict__ Vg,
             unsigned short* __restrict__ Y) {
  __shared__ unsigned short Kp[32 * DHEAD];
  __shared__ unsigned short Vp[32 * DHEAD];
  __shared__ unsigned short Ps[4][16 * 32];

  const int nqblk = TSEQ / 64;
  const int bh   = blockIdx.x / nqblk;
  const int qblk = blockIdx.x % nqblk;
  const int b = bh / N_HEAD, h = bh % N_HEAD;
  const int w    = threadIdx.x >> 5;
  const int lane = threadIdx.x & 31;
  const int half = lane >> 4, idx = lane & 15;
  const int tid  = threadIdx.x;

  const unsigned short* Qb = Q  + (size_t)bh * TSEQ * DHEAD;
  const unsigned short* Kb = Kg + (size_t)bh * TSEQ * DHEAD;
  const unsigned short* Vb = Vg + (size_t)bh * TSEQ * DHEAD;

  const int qrow0 = qblk * 64 + w * 16;
  v16bf qa0 = load_tile_rowk(Qb + (size_t)qrow0 * DHEAD, DHEAD);       // d 0..31
  v16bf qa1 = load_tile_rowk(Qb + (size_t)qrow0 * DHEAD + 32, DHEAD);  // d 32..63

  float mrow[8], lrow[8];
#pragma unroll
  for (int r = 0; r < 8; ++r) { mrow[r] = -INFINITY; lrow[r] = 0.f; }
  v8f acc0 = {}, acc1 = {}, acc2 = {}, acc3 = {};

  const int kend = (qblk + 1) * 64;
  for (int kv0 = 0; kv0 < kend; kv0 += 32) {
    __syncthreads();                       // protect Kp/Vp reuse
    {
      const unsigned short* Ksrc = Kb + (size_t)kv0 * DHEAD;  // 32x64 contiguous
      const unsigned short* Vsrc = Vb + (size_t)kv0 * DHEAD;
#if HAS_ASYNC_LDS
      GASYNC_B128(Ksrc + (size_t)tid * 8,         Kp + tid * 8);
      GASYNC_B128(Ksrc + (size_t)(tid + 128) * 8, Kp + (tid + 128) * 8);
      GASYNC_B128(Vsrc + (size_t)tid * 8,         Vp + tid * 8);
      GASYNC_B128(Vsrc + (size_t)(tid + 128) * 8, Vp + (tid + 128) * 8);
      __builtin_amdgcn_s_wait_asynccnt(0);
#else
      *(uint4*)&Kp[tid * 8]         = *(const uint4*)(Ksrc + (size_t)tid * 8);
      *(uint4*)&Kp[(tid + 128) * 8] = *(const uint4*)(Ksrc + (size_t)(tid + 128) * 8);
      *(uint4*)&Vp[tid * 8]         = *(const uint4*)(Vsrc + (size_t)tid * 8);
      *(uint4*)&Vp[(tid + 128) * 8] = *(const uint4*)(Vsrc + (size_t)(tid + 128) * 8);
#endif
    }
    __syncthreads();

    // S (16q x 32kv) as two 16x16 WMMA tiles, K-dim = D = 64 (2 chained WMMAs)
    v8f s0 = {}, s1 = {};
    s0 = wmma_bf16(qa0, load_tile_rowk(Kp, DHEAD), s0);
    s0 = wmma_bf16(qa1, load_tile_rowk(Kp + 32, DHEAD), s0);
    s1 = wmma_bf16(qa0, load_tile_rowk(Kp + 16 * DHEAD, DHEAD), s1);
    s1 = wmma_bf16(qa1, load_tile_rowk(Kp + 16 * DHEAD + 32, DHEAD), s1);

    // causal mask + row max (row lives across the 16 lanes of this half)
    float tmax[8];
#pragma unroll
    for (int r = 0; r < 8; ++r) {
      int row = qrow0 + r + half * 8;
      float v0 = (kv0 + idx      <= row) ? s0[r] : -INFINITY;
      float v1 = (kv0 + 16 + idx <= row) ? s1[r] : -INFINITY;
      s0[r] = v0; s1[r] = v1;
      tmax[r] = fmaxf(v0, v1);
    }
#pragma unroll
    for (int off = 1; off < 16; off <<= 1)
#pragma unroll
      for (int r = 0; r < 8; ++r)
        tmax[r] = fmaxf(tmax[r], __shfl_xor(tmax[r], off, 32));

    float alpha[8], psum[8];
#pragma unroll
    for (int r = 0; r < 8; ++r) {
      float mn = fmaxf(mrow[r], tmax[r]);
      alpha[r] = expf(mrow[r] - mn);
      float p0 = expf(s0[r] - mn);
      float p1 = expf(s1[r] - mn);
      mrow[r] = mn;
      psum[r] = p0 + p1;
      int prow = r + half * 8;
      Ps[w][prow * 32 + idx]      = f2bf(p0);   // P in row-major 16x32
      Ps[w][prow * 32 + 16 + idx] = f2bf(p1);
    }
#pragma unroll
    for (int r = 0; r < 8; ++r) {
#pragma unroll
      for (int off = 1; off < 16; off <<= 1)
        psum[r] += __shfl_xor(psum[r], off, 32);
      lrow[r] = lrow[r] * alpha[r] + psum[r];
    }
#pragma unroll
    for (int r = 0; r < 8; ++r) {
      acc0[r] *= alpha[r]; acc1[r] *= alpha[r];
      acc2[r] *= alpha[r]; acc3[r] *= alpha[r];
    }

    __syncthreads();                       // make Ps visible (layout shuffle)
    v16bf pa = load_tile_rowk(Ps[w], 32);  // P as A-operand (16x32)
    acc0 = wmma_bf16(pa, load_tile_colk(Vp +  0, DHEAD), acc0);
    acc1 = wmma_bf16(pa, load_tile_colk(Vp + 16, DHEAD), acc1);
    acc2 = wmma_bf16(pa, load_tile_colk(Vp + 32, DHEAD), acc2);
    acc3 = wmma_bf16(pa, load_tile_colk(Vp + 48, DHEAD), acc3);
  }

  v8f* accs[4] = {&acc0, &acc1, &acc2, &acc3};
#pragma unroll
  for (int t = 0; t < 4; ++t) {
#pragma unroll
    for (int r = 0; r < 8; ++r) {
      int row = qrow0 + r + half * 8;
      int d   = t * 16 + idx;
      float v = (*accs[t])[r] / lrow[r];
      Y[(size_t)(b * TSEQ + row) * EMB + h * DHEAD + d] = f2bf(v);
    }
  }
}

// ---------------- launcher ----------------

extern "C" void kernel_launch(void* const* d_in, const int* in_sizes, int n_in,
                              void* d_out, int out_size, void* d_ws, size_t ws_size,
                              hipStream_t stream) {
  (void)in_sizes; (void)n_in; (void)out_size; (void)ws_size;
  const float* x    = (const float*)d_in[0];
  const float* Wqkv = (const float*)d_in[1];
  const float* bqkv = (const float*)d_in[2];
  const float* Wout = (const float*)d_in[3];
  const float* bout = (const float*)d_in[4];
  float* out = (float*)d_out;

  const int M  = BATCH * TSEQ;   // 4096
  const int K  = EMB;            // 1024
  const int N3 = 3 * EMB;        // 3072

  char* ws = (char*)d_ws;
  size_t off = 0;
  auto wsalloc = [&](size_t bytes) -> char* {
    char* p = ws + off;
    off += (bytes + 255) & ~(size_t)255;
    return p;
  };
  unsigned short* xbf   = (unsigned short*)wsalloc((size_t)M * K * 2);
  unsigned short* WqkvT = (unsigned short*)wsalloc((size_t)N3 * K * 2);
  unsigned short* WoutT = (unsigned short*)wsalloc((size_t)K * K * 2);
  float*          qkv   = (float*)wsalloc((size_t)M * N3 * 4);
  unsigned short* Qh    = (unsigned short*)wsalloc((size_t)M * EMB * 2);
  unsigned short* Kh    = (unsigned short*)wsalloc((size_t)M * EMB * 2);
  unsigned short* Vh    = (unsigned short*)wsalloc((size_t)M * EMB * 2);
  unsigned short* Ybf   = (unsigned short*)wsalloc((size_t)M * EMB * 2);

  k_f32_to_bf16<<<(M * K + 255) / 256, 256, 0, stream>>>(x, xbf, M * K);
  k_transpose_bf16<<<dim3(N3 / 32, K / 32), dim3(32, 8), 0, stream>>>(Wqkv, WqkvT, K, N3);
  k_transpose_bf16<<<dim3(K / 32, K / 32), dim3(32, 8), 0, stream>>>(Wout, WoutT, K, K);

  k_gemm_bf16<<<dim3(M / 128, N3 / 128), 256, 0, stream>>>(xbf, WqkvT, bqkv, qkv, M, N3, K);

  int nrope = BATCH * N_HEAD * TSEQ * (DHEAD / 2);   // 2^21
  k_rope_pack<<<nrope / 256, 256, 0, stream>>>(qkv, Qh, Kh, Vh);

  k_flash_attn<<<BATCH * N_HEAD * (TSEQ / 64), 128, 0, stream>>>(Qh, Kh, Vh, Ybf);

  k_gemm_bf16<<<dim3(M / 128, EMB / 128), 256, 0, stream>>>(Ybf, WoutT, bout, out, M, EMB, K);
}
